// CrossAttention_82386062672424
// MI455X (gfx1250) — matrix operand order repaired and checked by reading, not compiled, not run on hardware
//
#include <hip/hip_runtime.h>

typedef __bf16 bf16;
typedef __attribute__((ext_vector_type(16))) __bf16 v16bf;
typedef __attribute__((ext_vector_type(8)))  float  v8f;
typedef __attribute__((ext_vector_type(4)))  int    v4i;

// ---------------------------------------------------------------------------
// gfx1250 async global->LDS staging (ASYNCcnt-tracked), guarded so the file
// still compiles if the builtins are not declared by this toolchain.
// Builtin signature (probe-confirmed by round-2 diagnostic):
//   void __builtin_amdgcn_global_load_async_to_lds_b128(v4i as(1)*, v4i as(3)*, imm off, imm cpol)
// ---------------------------------------------------------------------------
#if __has_builtin(__builtin_amdgcn_global_load_async_to_lds_b128) && \
    __has_builtin(__builtin_amdgcn_s_wait_asynccnt)
#define USE_ASYNC_LDS 1
#else
#define USE_ASYNC_LDS 0
#endif

#if USE_ASYNC_LDS
__device__ __forceinline__ void cp16_async(void* lds_dst, const void* gsrc) {
    __builtin_amdgcn_global_load_async_to_lds_b128(
        (__attribute__((address_space(1))) v4i*)gsrc,
        (__attribute__((address_space(3))) v4i*)lds_dst, 0, 0);
}
#endif

// Stage `rows` rows of 1024 bytes (512 bf16) from global into LDS (pitch in bytes).
__device__ __forceinline__ void stage_rows(void* lds, const void* gbase, int rows,
                                           int ldsPitchB, int tid) {
#if USE_ASYNC_LDS
    for (int u = tid; u < rows * 64; u += 128) {          // 16B units
        int r = u >> 6, c = u & 63;
        cp16_async((char*)lds + r * ldsPitchB + c * 16,
                   (const char*)gbase + (size_t)r * 1024 + c * 16);
    }
    __builtin_amdgcn_s_wait_asynccnt(0);
#else
    const unsigned int* src = (const unsigned int*)gbase;
    for (int idx = tid; idx < rows * 256; idx += 128) {   // dwords
        int r = idx >> 8, c = idx & 255;
        *(unsigned int*)((char*)lds + r * ldsPitchB + c * 4) = src[idx];
    }
#endif
}

// ---------------------------------------------------------------------------
// Wave32 WMMA fragment loaders (16x16x32 bf16, contiguous k order in A and B: k = 16*(lane/16) + element)
// ---------------------------------------------------------------------------

// A (16x32, MxK): lanes 0-15 -> M=lane, K in {0..15}; lanes 16-31 -> K in {16..31}
__device__ __forceinline__ v16bf load_a_frag(const bf16* As, int lane, int pitch) {
    int m  = lane & 15;
    int kb = (lane < 16) ? 0 : 16;
    const bf16* row = As + (size_t)m * pitch;
    v16bf a;
#pragma unroll
    for (int i = 0; i < 8; ++i) {
        int k = kb + 2 * i;
        a[2 * i]     = row[k];
        a[2 * i + 1] = row[k + 1];
    }
    return a;
}

// B (32x16, KxN) from row-major [K][N] storage
__device__ __forceinline__ v16bf load_b_frag(const bf16* Bs, int lane, int n0, int pitch) {
    int n  = n0 + (lane & 15);
    int kb = (lane < 16) ? 0 : 16;
    v16bf b;
#pragma unroll
    for (int i = 0; i < 16; ++i)
        b[i] = Bs[(size_t)(kb + i) * pitch + n];
    return b;
}

// B (32x16, KxN) from transposed [N][K] storage (contraction contiguous -> ds_load_b128)
__device__ __forceinline__ v16bf load_bt_frag(const bf16* Bs, int lane, int n0, int pitch) {
    int n  = n0 + (lane & 15);
    int kb = (lane < 16) ? 0 : 16;
    const bf16* row = Bs + (size_t)n * pitch;
    v16bf b;
#pragma unroll
    for (int i = 0; i < 16; ++i)
        b[i] = row[kb + i];
    return b;
}

// ---------------------------------------------------------------------------
// Generic tiled WMMA GEMM:  out[b] = A[b] (MxK) @ W (KxN) + bias  (+ epilogue)
//   block = 128 threads = 4 waves; block tile 64x64, K-step 32
// ---------------------------------------------------------------------------

enum { A_F32_ROW = 0, A_F32_TCOL = 1, A_BF16_ROW = 2 };
enum { E_STORE_BF16 = 0, E_ADDX_TRANS_F32 = 1 };

#define BM  64
#define BN  64
#define BKK 32
#define AP  40   // A tile pitch (bf16), 80B rows: 16B-aligned for b128
#define BPt 40   // transposed B tile pitch (bf16)
#define OPI 68   // output staging pitch (f32)

template <int AM, int EM>
__global__ __launch_bounds__(128)
void gemm_wmma(const void* __restrict__ Agv, int lda, unsigned long long strideA,
               const float* __restrict__ W, const float* __restrict__ bias,
               bf16* __restrict__ Ob, unsigned long long strideO,
               const float* __restrict__ Xres, float* __restrict__ Fout,
               unsigned long long strideX, int ldx,
               int M, int N, int K)
{
    __shared__ bf16  Abuf[BM * AP];       // [M][K]
    __shared__ bf16  Bbuf[BN * BPt];      // [N][K] (transposed)
    __shared__ float Obuf[BN * OPI];      // [N][M] staging for transposed store

    const int tid  = threadIdx.x;
    const int lane = tid & 31;
    const int wv   = tid >> 5;
    const int ntn  = N / BN;
    const int m0   = (blockIdx.x / ntn) * BM;
    const int n0   = (blockIdx.x % ntn) * BN;
    const int b    = blockIdx.y;

    const float* Af = (const float*)Agv + (size_t)b * strideA;
    const bf16*  Ab = (const bf16*)Agv  + (size_t)b * strideA;

    v8f zero = {0.f,0.f,0.f,0.f,0.f,0.f,0.f,0.f};
    v8f acc[4];
#pragma unroll
    for (int t = 0; t < 4; ++t) acc[t] = zero;

    for (int k0 = 0; k0 < K; k0 += BKK) {
        __syncthreads();
        // ---- stage A tile (64x32) ----
        if (AM == A_F32_TCOL) {
#pragma unroll
            for (int i = 0; i < 16; ++i) {
                int idx = i * 128 + tid;
                int r = idx & 63, c = idx >> 6;               // consecutive tid -> consecutive m
                Abuf[r * AP + c] = (bf16)Af[(size_t)(k0 + c) * lda + (m0 + r)];
            }
        } else if (AM == A_F32_ROW) {
#pragma unroll
            for (int i = 0; i < 16; ++i) {
                int idx = i * 128 + tid;
                int c = idx & 31, r = idx >> 5;               // consecutive tid -> consecutive k
                Abuf[r * AP + c] = (bf16)Af[(size_t)(m0 + r) * lda + k0 + c];
            }
        } else { // A_BF16_ROW
#if USE_ASYNC_LDS
            for (int u = tid; u < 64 * 4; u += 128) {         // 64 rows x 4 x 16B
                int r = u >> 2, c = u & 3;
                cp16_async((char*)Abuf + r * (AP * 2) + c * 16,
                           (const char*)(Ab + (size_t)(m0 + r) * lda + k0) + c * 16);
            }
#else
            for (int idx = tid; idx < 64 * 16; idx += 128) {  // dwords, 16/row
                int r = idx >> 4, c = idx & 15;
                *(unsigned int*)((char*)Abuf + r * (AP * 2) + c * 4) =
                    ((const unsigned int*)(Ab + (size_t)(m0 + r) * lda + k0))[c];
            }
#endif
        }
        // ---- stage B tile (32x64 of W, stored transposed [n][k]) ----
#pragma unroll
        for (int i = 0; i < 16; ++i) {
            int idx = i * 128 + tid;
            int c = idx & 63, r = idx >> 6;                   // coalesced read of W row
            Bbuf[c * BPt + r] = (bf16)W[(size_t)(k0 + r) * N + n0 + c];
        }
#if USE_ASYNC_LDS
        if (AM == A_BF16_ROW) __builtin_amdgcn_s_wait_asynccnt(0);
#endif
        __syncthreads();

        v16bf a = load_a_frag(Abuf + wv * 16 * AP, lane, AP);
#pragma unroll
        for (int t = 0; t < 4; ++t) {
            v16bf bb = load_bt_frag(Bbuf, lane, t * 16, BPt);
            acc[t] = __builtin_amdgcn_wmma_f32_16x16x32_bf16(
                         false, a, false, bb, (short)0, acc[t], false, false);
        }
    }

    // ---- epilogue ----
    const int mb = (lane < 16) ? 0 : 8;
    if (EM == E_STORE_BF16) {
#pragma unroll
        for (int t = 0; t < 4; ++t) {
            int   ng = n0 + t * 16 + (lane & 15);
            float bs = bias[ng];
#pragma unroll
            for (int r = 0; r < 8; ++r) {
                int mg = m0 + wv * 16 + r + mb;
                Ob[(size_t)b * strideO + (size_t)mg * N + ng] = (bf16)(acc[t][r] + bs);
            }
        }
    } else {
        // transpose through LDS, then coalesced residual+store: out[b][n][m] = x + v
        __syncthreads();
#pragma unroll
        for (int t = 0; t < 4; ++t) {
            int   nl = t * 16 + (lane & 15);
            float bs = bias[n0 + nl];
#pragma unroll
            for (int r = 0; r < 8; ++r) {
                int ml = wv * 16 + r + mb;
                Obuf[nl * OPI + ml] = acc[t][r] + bs;
            }
        }
        __syncthreads();
        for (int idx = tid; idx < 64 * 64; idx += 128) {
            int rn = idx >> 6, cm = idx & 63;
            size_t xi = (size_t)b * strideX + (size_t)(n0 + rn) * ldx + (m0 + cm);
            Fout[xi] = Xres[xi] + Obuf[rn * OPI + cm];
        }
    }
}

// ---------------------------------------------------------------------------
// Attention: per block (4 waves, 128 thr) handle 32 queries x all 512 keys.
//   ~130 KB LDS/WG -> 2 workgroups per 320 KB WGP.
// ---------------------------------------------------------------------------

#define QTILE 32
#define KCH   32    // keys per streamed chunk
#define DH    512
#define QP    520   // Q/P pitch (bf16), 1040B rows (16B aligned)
#define SP    516   // S pitch (f32)
#define KP    520   // K/V chunk pitch (bf16)

__global__ __launch_bounds__(128)
void attn_wmma(const bf16* __restrict__ qb, const bf16* __restrict__ kb,
               const bf16* __restrict__ vb, bf16* __restrict__ ob)
{
    extern __shared__ char smem[];
    bf16*  Qs     = (bf16*)smem;                               // 32x520 bf16 (later: P)
    float* Ss     = (float*)(smem + QTILE * QP * 2);           // 32x516 f32
    bf16*  KVs    = (bf16*)((char*)Ss + QTILE * SP * 4);       // 32x520 bf16
    float* red    = (float*)((char*)KVs + KCH * KP * 2);       // 128 f32
    float* rowmax = red + 128;                                 // 32 f32
    float* rowinv = rowmax + 32;                               // 32 f32

    const int tid  = threadIdx.x;
    const int lane = tid & 31;
    const int wv   = tid >> 5;
    const int b    = blockIdx.y;
    const int q0   = blockIdx.x * QTILE;

    const size_t qoff  = (size_t)b * 4096 * DH + (size_t)q0 * DH;
    const size_t kvoff = (size_t)b * 512 * DH;

    // ---- stage Q tile: 32 rows x 512 bf16 ----
    stage_rows(Qs, qb + qoff, QTILE, QP * 2, tid);

    // ---- phase 1: S = (Q K^T) * scale, streamed over 16 chunks of 32 keys ----
    const float scl = 0.044194173824159216f; // 512^-0.5
    const int   srt = (wv & 1) * 16;         // wave's query-row tile
    const int   scb = (wv >> 1) * 16;        // wave's key-col base within chunk
    const int   mb  = (lane < 16) ? 0 : 8;

    for (int kc = 0; kc < 16; ++kc) {
        __syncthreads();
        stage_rows(KVs, kb + kvoff + (size_t)kc * KCH * DH, KCH, KP * 2, tid);
        if (kc + 1 < 16)   // gfx1250 global_prefetch_b8 for next chunk (32KB)
            __builtin_prefetch((const char*)(kb + kvoff +
                               (size_t)(kc + 1) * KCH * DH) + tid * 256, 0, 1);
        __syncthreads();

        v8f s0 = {0.f,0.f,0.f,0.f,0.f,0.f,0.f,0.f};
#pragma unroll
        for (int ks = 0; ks < 16; ++ks) {
            v16bf a  = load_a_frag(Qs + srt * QP + ks * 32, lane, QP);
            v16bf b0 = load_bt_frag(KVs + ks * 32, lane, scb, KP);
            s0 = __builtin_amdgcn_wmma_f32_16x16x32_bf16(false, a, false, b0,
                                                         (short)0, s0, false, false);
        }
#pragma unroll
        for (int r = 0; r < 8; ++r) {
            int m = srt + r + mb;
            Ss[m * SP + kc * 32 + scb + (lane & 15)] = s0[r] * scl;
        }
    }
    __syncthreads();

    // ---- phase 2: exact softmax over 512 keys; P (bf16) overwrites Q buffer ----
    {
        int r = tid & 31, seg = tid >> 5;
        const float* srow = Ss + r * SP + seg * 128;
        float mx = -3.4e38f;
#pragma unroll 4
        for (int j = 0; j < 128; ++j) mx = fmaxf(mx, srow[j]);
        red[seg * 32 + r] = mx;
        __syncthreads();
        if (tid < 32)
            rowmax[tid] = fmaxf(fmaxf(red[tid], red[32 + tid]),
                                fmaxf(red[64 + tid], red[96 + tid]));
        __syncthreads();
        float m = rowmax[r];
        bf16* prow = Qs + r * QP + seg * 128;   // Q no longer needed -> P
        float s = 0.f;
#pragma unroll 4
        for (int j = 0; j < 128; ++j) {
            float e = __expf(srow[j] - m);
            prow[j] = (bf16)e;
            s += e;
        }
        red[seg * 32 + r] = s;
        __syncthreads();
        if (tid < 32)
            rowinv[tid] = 1.0f / (red[tid] + red[32 + tid] + red[64 + tid] + red[96 + tid]);
        __syncthreads();
    }

    // ---- phase 3: O = P @ V, two 256-column passes, V streamed in 32-key chunks ----
    const size_t obase = (size_t)b * 4096 * DH + (size_t)q0 * DH;
    for (int cp = 0; cp < 2; ++cp) {
        v8f z = {0.f,0.f,0.f,0.f,0.f,0.f,0.f,0.f};
        v8f oa[2][4];
#pragma unroll
        for (int rt = 0; rt < 2; ++rt)
#pragma unroll
            for (int t = 0; t < 4; ++t) oa[rt][t] = z;

        const int nc0 = cp * 256 + wv * 64;
        for (int kc = 0; kc < 16; ++kc) {
            __syncthreads();
            stage_rows(KVs, vb + kvoff + (size_t)kc * KCH * DH, KCH, KP * 2, tid);
            if (kc + 1 < 16)
                __builtin_prefetch((const char*)(vb + kvoff +
                                   (size_t)(kc + 1) * KCH * DH) + tid * 256, 0, 1);
            __syncthreads();

            v16bf a0 = load_a_frag(Qs + kc * 32,           lane, QP);
            v16bf a1 = load_a_frag(Qs + 16 * QP + kc * 32, lane, QP);
#pragma unroll
            for (int t = 0; t < 4; ++t) {
                v16bf bb = load_b_frag(KVs, lane, nc0 + t * 16, KP);
                oa[0][t] = __builtin_amdgcn_wmma_f32_16x16x32_bf16(
                               false, a0, false, bb, (short)0, oa[0][t], false, false);
                oa[1][t] = __builtin_amdgcn_wmma_f32_16x16x32_bf16(
                               false, a1, false, bb, (short)0, oa[1][t], false, false);
            }
        }
#pragma unroll
        for (int rt = 0; rt < 2; ++rt)
#pragma unroll
            for (int t = 0; t < 4; ++t) {
                int ng = nc0 + t * 16 + (lane & 15);
#pragma unroll
                for (int r = 0; r < 8; ++r) {
                    int m = rt * 16 + r + mb;
                    ob[obase + (size_t)m * DH + ng] = (bf16)(oa[rt][t][r] * rowinv[m]);
                }
            }
    }
}

// ---------------------------------------------------------------------------
// Host launch
// ---------------------------------------------------------------------------

extern "C" void kernel_launch(void* const* d_in, const int* in_sizes, int n_in,
                              void* d_out, int out_size, void* d_ws, size_t ws_size,
                              hipStream_t stream) {
    (void)in_sizes; (void)n_in; (void)out_size; (void)ws_size;

    const float* x   = (const float*)d_in[0];  // [8,512,64,64]
    const float* ctx = (const float*)d_in[1];  // [8,512,768]
    const float* Wq  = (const float*)d_in[2];
    const float* bq  = (const float*)d_in[3];
    const float* Wk  = (const float*)d_in[4];
    const float* bk  = (const float*)d_in[5];
    const float* Wv  = (const float*)d_in[6];
    const float* bv  = (const float*)d_in[7];
    const float* Wo  = (const float*)d_in[8];
    const float* bo  = (const float*)d_in[9];
    float* out = (float*)d_out;

    const int B = 8, C = 512, HW = 4096, CTX = 768, LC = 512;

    bf16* qb = (bf16*)d_ws;                              // [B,4096,512] bf16
    bf16* kb = qb + (size_t)B * HW * C;                  // [B, 512,512] bf16
    bf16* vb = kb + (size_t)B * LC * C;                  // [B, 512,512] bf16
    bf16* ob = vb + (size_t)B * LC * C;                  // [B,4096,512] bf16

    dim3 blk(128);

    // Q = xr @ Wq + bq   (xr = x viewed [HW,C], i.e. transposed read of x)
    gemm_wmma<A_F32_TCOL, E_STORE_BF16>
        <<<dim3((HW / 64) * (C / 64), B), blk, 0, stream>>>(
            x, /*lda=*/HW, /*strideA=*/(unsigned long long)C * HW,
            Wq, bq, qb, (unsigned long long)HW * C,
            nullptr, nullptr, 0ull, 0, HW, C, C);

    // K = context @ Wk + bk
    gemm_wmma<A_F32_ROW, E_STORE_BF16>
        <<<dim3((LC / 64) * (C / 64), B), blk, 0, stream>>>(
            ctx, /*lda=*/CTX, (unsigned long long)LC * CTX,
            Wk, bk, kb, (unsigned long long)LC * C,
            nullptr, nullptr, 0ull, 0, LC, C, CTX);

    // V = context @ Wv + bv
    gemm_wmma<A_F32_ROW, E_STORE_BF16>
        <<<dim3((LC / 64) * (C / 64), B), blk, 0, stream>>>(
            ctx, /*lda=*/CTX, (unsigned long long)LC * CTX,
            Wv, bv, vb, (unsigned long long)LC * C,
            nullptr, nullptr, 0ull, 0, LC, C, CTX);

    // Attention: softmax(Q K^T / sqrt(C)) @ V
    size_t smem = (size_t)QTILE * QP * 2 + (size_t)QTILE * SP * 4 +
                  (size_t)KCH * KP * 2 + (128 + 64) * sizeof(float);
    attn_wmma<<<dim3(HW / QTILE, B), blk, smem, stream>>>(qb, kb, vb, ob);

    // out = x + (O @ Wo + bo) with transposed store back to [B,C,H,W]
    gemm_wmma<A_BF16_ROW, E_ADDX_TRANS_F32>
        <<<dim3((HW / 64) * (C / 64), B), blk, 0, stream>>>(
            ob, /*lda=*/C, (unsigned long long)HW * C,
            Wo, bo, nullptr, 0ull,
            x, out, (unsigned long long)C * HW, /*ldx=*/HW,
            HW, C, C);
}
